// SwinTransformerBlock3D_40991167873321
// MI455X (gfx1250) — compile-verified, matrix operand order checked by reading, not run on hardware
//
#include <hip/hip_runtime.h>
#include <hip/hip_bf16.h>
#include <math.h>

// ---------------------------------------------------------------------------
// Swin-3D block on gfx1250: all GEMMs + attention via v_wmma_f32_16x16x32_f16
// GEMM register-blocked 32x48 per wave: 6 WMMAs per 10 b128 loads per K-step.
// ---------------------------------------------------------------------------

typedef __attribute__((ext_vector_type(16))) _Float16 v16h;
typedef __attribute__((ext_vector_type(8)))  _Float16 h8;
typedef __attribute__((ext_vector_type(8)))  float    v8f;

#define NWIN   256          // windows (4*8*8)
#define NTOK   196          // tokens per window (4*7*7)
#define NPAD   224          // padded tokens (14 * 16, divisible by 32)
#define HDPAD  32           // padded head dim (12 -> 32)
#define CH     96
#define MLPD   384
#define TTOT   50176        // NWIN*NTOK
#define QSCALE 0.28867513459481287f   // 12^-0.5

static __device__ inline v16h cat8(h8 a, h8 b) {
  return __builtin_shufflevector(a, b, 0,1,2,3,4,5,6,7,8,9,10,11,12,13,14,15);
}

// ---------------------------------------------------------------- zero fill
__global__ void fillz_kernel(unsigned int* p, size_t n) {
  size_t i  = (size_t)blockIdx.x * blockDim.x + threadIdx.x;
  size_t st = (size_t)gridDim.x * blockDim.x;
  for (; i < n; i += st) p[i] = 0u;
}

// --------------------------------------------- weight convert fp32 -> f16^T
// dst[n*K + k] = src[k*N + n]   (N-major, K-contiguous: B-fragment friendly)
__global__ void cvt_t_kernel(const float* __restrict__ src,
                             _Float16* __restrict__ dst, int K, int N) {
  int i = blockIdx.x * 256 + threadIdx.x;
  if (i >= K * N) return;
  int k = i / N, n = i % N;
  dst[(size_t)n * K + k] = (_Float16)src[i];
}

// ----------------------------------------------------- padded bias(+pad-inf)
// bp[h][i][j], i,j in [0,224): rel-pos bias; padded cols get -1e30 (softmax 0)
__global__ void biaspad_kernel(const float* __restrict__ rpb,
                               const int* __restrict__ rel,
                               float* __restrict__ bp) {
  int i = blockIdx.x * 256 + threadIdx.x;
  if (i >= 8 * NPAD * NPAD) return;
  int h = i / (NPAD * NPAD);
  int r = (i / NPAD) % NPAD;
  int c = i % NPAD;
  float v;
  if (c >= NTOK)      v = -1e30f;
  else if (r >= NTOK) v = 0.0f;
  else                v = rpb[rel[r * NTOK + c] * 8 + h];
  bp[i] = v;
}

// ----------------------------------------------------------- LayerNorm (f16)
// shifted=1: token index is window-order, gathers from x with roll(-2,-3,-3)
// shifted=0: identity mapping (natural order)
__global__ void ln_kernel(const float* __restrict__ in,
                          const float* __restrict__ g,
                          const float* __restrict__ b,
                          _Float16* __restrict__ out, int shifted) {
  int t    = blockIdx.x * 4 + (threadIdx.x >> 5);
  int lane = threadIdx.x & 31;
  size_t src;
  if (shifted) {
    int w  = t / NTOK, nt = t % NTOK;
    int dw = w >> 6, hw = (w >> 3) & 7, ww = w & 7;
    int od = nt / 49, oh = (nt / 7) % 7, ow = nt % 7;
    int dd = (dw * 4 + od + 2) & 15;
    int hh = (hw * 7 + oh + 3) % 56;
    int w2 = (ww * 7 + ow + 3) % 56;
    src = ((size_t)(dd * 56 + hh) * 56 + w2) * CH;
  } else {
    src = (size_t)t * CH;
  }
  float v0 = in[src + lane];
  float v1 = in[src + lane + 32];
  float v2 = in[src + lane + 64];
  float s = v0 + v1 + v2;
  float q = v0 * v0 + v1 * v1 + v2 * v2;
  for (int i = 16; i > 0; i >>= 1) {
    s += __shfl_xor(s, i, 32);
    q += __shfl_xor(q, i, 32);
  }
  float mean = s * (1.0f / CH);
  float var  = q * (1.0f / CH) - mean * mean;
  float rstd = rsqrtf(var + 1e-5f);
  size_t o = (size_t)t * CH;
  out[o + lane]      = (_Float16)((v0 - mean) * rstd * g[lane]      + b[lane]);
  out[o + lane + 32] = (_Float16)((v1 - mean) * rstd * g[lane + 32] + b[lane + 32]);
  out[o + lane + 64] = (_Float16)((v2 - mean) * rstd * g[lane + 64] + b[lane + 64]);
}

// ------------------------------------------------------- generic WMMA GEMM
// One wave computes a 32x48 tile of C = A(MxK,f16,row-major) * Bt^T:
// 2 A-fragments x 3 B-fragments -> 6 accumulators, 6 WMMAs per K=32 step.
// Bt is stored N-major / K-contiguous. mode selects the fused epilogue.
__global__ void __launch_bounds__(32)
gemm_wmma_kernel(const _Float16* __restrict__ A,
                 const _Float16* __restrict__ Bt,
                 const float* __restrict__ bias, int K, int mode,
                 _Float16* __restrict__ qh, _Float16* __restrict__ kh,
                 _Float16* __restrict__ vt,
                 const float* __restrict__ xsrc, float* __restrict__ yout,
                 _Float16* __restrict__ gout,
                 const float* __restrict__ yin, float* __restrict__ dout) {
  int lane = threadIdx.x;
  int bn = blockIdx.x, bm = blockIdx.y;
  int khi = (lane >> 4) * 8;
  const _Float16* arow0 = A  + (size_t)(bm * 32 + (lane & 15)) * K;
  const _Float16* arow1 = arow0 + (size_t)16 * K;
  const _Float16* brow0 = Bt + (size_t)(bn * 48 + (lane & 15)) * K;
  const _Float16* brow1 = brow0 + (size_t)16 * K;
  const _Float16* brow2 = brow0 + (size_t)32 * K;
  v8f acc[2][3] = {};
  for (int s = 0; s < K; s += 32) {
    v16h a0 = cat8(*(const h8*)(arow0 + s + khi), *(const h8*)(arow0 + s + khi + 16));
    v16h a1 = cat8(*(const h8*)(arow1 + s + khi), *(const h8*)(arow1 + s + khi + 16));
    v16h b0 = cat8(*(const h8*)(brow0 + s + khi), *(const h8*)(brow0 + s + khi + 16));
    v16h b1 = cat8(*(const h8*)(brow1 + s + khi), *(const h8*)(brow1 + s + khi + 16));
    v16h b2 = cat8(*(const h8*)(brow2 + s + khi), *(const h8*)(brow2 + s + khi + 16));
    acc[0][0] = __builtin_amdgcn_wmma_f32_16x16x32_f16(false, a0, false, b0, (short)0, acc[0][0], false, false);
    acc[0][1] = __builtin_amdgcn_wmma_f32_16x16x32_f16(false, a0, false, b1, (short)0, acc[0][1], false, false);
    acc[0][2] = __builtin_amdgcn_wmma_f32_16x16x32_f16(false, a0, false, b2, (short)0, acc[0][2], false, false);
    acc[1][0] = __builtin_amdgcn_wmma_f32_16x16x32_f16(false, a1, false, b0, (short)0, acc[1][0], false, false);
    acc[1][1] = __builtin_amdgcn_wmma_f32_16x16x32_f16(false, a1, false, b1, (short)0, acc[1][1], false, false);
    acc[1][2] = __builtin_amdgcn_wmma_f32_16x16x32_f16(false, a1, false, b2, (short)0, acc[1][2], false, false);
  }
#pragma unroll
  for (int im = 0; im < 2; im++) {
#pragma unroll
    for (int jn = 0; jn < 3; jn++) {
      int ncol = bn * 48 + jn * 16 + (lane & 15);
      float bv = bias[ncol];
#pragma unroll
      for (int jv = 0; jv < 8; jv++) {
        int row = bm * 32 + im * 16 + jv + ((lane >> 4) << 3);
        float val = acc[im][jn][jv] + bv;
        if (mode == 0) {                  // QKV scatter into padded q/k/v^T
          int which = ncol / CH, r = ncol % CH;
          int h = r / 12, hd = r % 12;
          int w = row / NTOK, nt = row % NTOK;
          size_t idx = ((size_t)(w * 8 + h) * NPAD + nt) * HDPAD + hd;
          if (which == 0)      qh[idx] = (_Float16)(val * QSCALE);
          else if (which == 1) kh[idx] = (_Float16)val;
          else vt[((size_t)(w * 8 + h) * 16 + hd) * NPAD + nt] = (_Float16)val;
        } else if (mode == 1) {           // proj + window-reverse + roll + res
          int w  = row / NTOK, nt = row % NTOK;
          int dw = w >> 6, hw = (w >> 3) & 7, ww = w & 7;
          int od = nt / 49, oh = (nt / 7) % 7, ow = nt % 7;
          int dd = (dw * 4 + od + 2) & 15;
          int hh = (hw * 7 + oh + 3) % 56;
          int w2 = (ww * 7 + ow + 3) % 56;
          size_t nat = ((size_t)(dd * 56 + hh) * 56 + w2) * CH + ncol;
          yout[nat] = xsrc[nat] + val;
        } else if (mode == 2) {           // fc1 + exact GELU
          float ge = 0.5f * val * (1.0f + erff(val * 0.70710678118654752f));
          gout[(size_t)row * MLPD + ncol] = (_Float16)ge;
        } else {                          // fc2 + residual -> output
          size_t i2 = (size_t)row * CH + ncol;
          dout[i2] = val + yin[i2];
        }
      }
    }
  }
}

// ----------------------------------------------------------- fused attention
// One wave per (window, head, 16-query-row block).
__global__ void __launch_bounds__(32)
attn_kernel(const _Float16* __restrict__ Q, const _Float16* __restrict__ Kb,
            const _Float16* __restrict__ Vt, const float* __restrict__ bp,
            const float* __restrict__ mask, _Float16* __restrict__ O) {
  __shared__ __align__(16) float    S[16 * NPAD];
  __shared__ __align__(16) _Float16 P[16 * NPAD];
  int lane = threadIdx.x;
  int wh = blockIdx.y;           // window*8 + head
  int w  = wh >> 3, h = wh & 7;
  int rb = blockIdx.x;           // 16-row block, 0..12
  const _Float16* qb = Q  + (size_t)wh * NPAD * HDPAD;
  const _Float16* kb = Kb + (size_t)wh * NPAD * HDPAD;
  const _Float16* vb = Vt + (size_t)wh * 16 * NPAD;
  int khi  = (lane >> 4) * 8;
  int mrow = rb * 16 + (lane & 15);

  h8 ql = *(const h8*)(qb + mrow * HDPAD + khi);
  h8 qh = *(const h8*)(qb + mrow * HDPAD + khi + 16);
  v16h aq = cat8(ql, qh);

  for (int jt = 0; jt < NPAD / 16; jt++) {       // 14 column tiles
    int col = jt * 16 + (lane & 15);
    h8 kl = *(const h8*)(kb + col * HDPAD + khi);
    h8 kh = *(const h8*)(kb + col * HDPAD + khi + 16);
    v8f c = {};
    c = __builtin_amdgcn_wmma_f32_16x16x32_f16(false, aq, false, cat8(kl, kh),
                                               (short)0, c, false, false);
#pragma unroll
    for (int jv = 0; jv < 8; jv++) {
      int mm  = jv + ((lane >> 4) << 3);
      int row = rb * 16 + mm;
      int cc  = jt * 16 + (lane & 15);
      float val = c[jv] + bp[((size_t)h * NPAD + row) * NPAD + cc];
      if (row < NTOK && cc < NTOK)
        val += mask[((size_t)w * NTOK + row) * NTOK + cc];
      S[mm * NPAD + cc] = val;
    }
  }
  __syncthreads();
  if (lane < 16) {
    float mx = -1e30f;
    for (int c = 0; c < NPAD; c++) mx = fmaxf(mx, S[lane * NPAD + c]);
    float sum = 0.0f;
    for (int c = 0; c < NPAD; c++) {
      float e = __expf(S[lane * NPAD + c] - mx);
      sum += e;
      S[lane * NPAD + c] = e;
    }
    float inv = 1.0f / sum;
    for (int c = 0; c < NPAD; c++)
      P[lane * NPAD + c] = (_Float16)(S[lane * NPAD + c] * inv);
  }
  __syncthreads();

  v8f o = {};
  for (int s = 0; s < NPAD / 32; s++) {          // 7 K-steps over tokens
    int k0 = s * 32 + khi;
    h8 pl = *(const h8*)(P + (lane & 15) * NPAD + k0);
    h8 ph = *(const h8*)(P + (lane & 15) * NPAD + k0 + 16);
    h8 vl = *(const h8*)(vb + (lane & 15) * NPAD + k0);
    h8 vh = *(const h8*)(vb + (lane & 15) * NPAD + k0 + 16);
    o = __builtin_amdgcn_wmma_f32_16x16x32_f16(false, cat8(pl, ph),
                                               false, cat8(vl, vh),
                                               (short)0, o, false, false);
  }
#pragma unroll
  for (int jv = 0; jv < 8; jv++) {
    int mm  = jv + ((lane >> 4) << 3);
    int row = rb * 16 + mm;
    int hd  = lane & 15;
    if (row < NTOK && hd < 12)
      O[((size_t)w * NTOK + row) * CH + h * 12 + hd] = (_Float16)o[jv];
  }
}

// ---------------------------------------------------------------------------
extern "C" void kernel_launch(void* const* d_in, const int* in_sizes, int n_in,
                              void* d_out, int out_size, void* d_ws, size_t ws_size,
                              hipStream_t stream) {
  const float* x      = (const float*)d_in[0];
  const float* g1     = (const float*)d_in[1];
  const float* b1     = (const float*)d_in[2];
  const float* qkv_w  = (const float*)d_in[3];
  const float* qkv_b  = (const float*)d_in[4];
  const float* proj_w = (const float*)d_in[5];
  const float* proj_b = (const float*)d_in[6];
  const float* rpb    = (const float*)d_in[7];
  const float* g2     = (const float*)d_in[8];
  const float* b2     = (const float*)d_in[9];
  const float* fc1_w  = (const float*)d_in[10];
  const float* fc1_b  = (const float*)d_in[11];
  const float* fc2_w  = (const float*)d_in[12];
  const float* fc2_b  = (const float*)d_in[13];
  const int*   rel    = (const int*)d_in[14];
  const float* mask   = (const float*)d_in[15];
  float* out = (float*)d_out;

  char* ws = (char*)d_ws;
  size_t off = 0;
  _Float16* Qh = (_Float16*)(ws + off); off += (size_t)NWIN * 8 * NPAD * HDPAD * 2;  // 29.36 MB
  _Float16* Kh = (_Float16*)(ws + off); off += (size_t)NWIN * 8 * NPAD * HDPAD * 2;  // 29.36 MB
  _Float16* Vt = (_Float16*)(ws + off); off += (size_t)NWIN * 8 * 16 * NPAD * 2;     // 14.68 MB
  size_t zfill_u32 = ((size_t)NWIN * 8 * (NPAD * HDPAD * 2 + 16 * NPAD) * 2) / 4;    // q+k+v
  float*    BP = (float*)(ws + off);    off += (size_t)8 * NPAD * NPAD * 4;          // 1.6 MB
  _Float16* XW = (_Float16*)(ws + off); off += (size_t)TTOT * CH * 2;   // LN1 out, reused as attn O
  float*    Y  = (float*)(ws + off);    off += (size_t)TTOT * CH * 4;   // residual-1 result
  _Float16* WQ = (_Float16*)(ws + off); off += (size_t)CH * 288 * 2;
  _Float16* WP = (_Float16*)(ws + off); off += (size_t)CH * CH * 2;
  _Float16* W1 = (_Float16*)(ws + off); off += (size_t)CH * MLPD * 2;
  _Float16* W2 = (_Float16*)(ws + off); off += (size_t)MLPD * CH * 2;
  _Float16* G  = Qh;   // mlp hidden (38.5 MB) aliases q+k region (58.7 MB)
  _Float16* H2 = Vt;   // LN2 out (9.6 MB) aliases v region (14.7 MB)

  // 1) zero the padded q/k/v buffers (pads make WMMA K-steps exact)
  fillz_kernel<<<2048, 256, 0, stream>>>((unsigned int*)Qh, zfill_u32);
  // 2) weights -> f16 transposed
  cvt_t_kernel<<<(CH * 288 + 255) / 256, 256, 0, stream>>>(qkv_w, WQ, CH, 288);
  cvt_t_kernel<<<(CH * CH + 255) / 256, 256, 0, stream>>>(proj_w, WP, CH, CH);
  cvt_t_kernel<<<(CH * MLPD + 255) / 256, 256, 0, stream>>>(fc1_w, W1, CH, MLPD);
  cvt_t_kernel<<<(MLPD * CH + 255) / 256, 256, 0, stream>>>(fc2_w, W2, MLPD, CH);
  // 3) padded relative-position bias table
  biaspad_kernel<<<(8 * NPAD * NPAD + 255) / 256, 256, 0, stream>>>(rpb, rel, BP);
  // 4) LN1 + cyclic shift + window partition -> f16
  ln_kernel<<<TTOT / 4, 128, 0, stream>>>(x, g1, b1, XW, 1);
  // 5) QKV GEMM (M=50176, K=96, N=288) with q/k/v scatter epilogue
  gemm_wmma_kernel<<<dim3(288 / 48, TTOT / 32), 32, 0, stream>>>(
      XW, WQ, qkv_b, CH, 0, Qh, Kh, Vt, nullptr, nullptr, nullptr, nullptr, nullptr);
  // 6) fused attention -> O (reuses XW buffer)
  attn_kernel<<<dim3(13, NWIN * 8), 32, 0, stream>>>(Qh, Kh, Vt, BP, mask, XW);
  // 7) proj GEMM + window-reverse + roll-back + residual -> Y
  gemm_wmma_kernel<<<dim3(CH / 48, TTOT / 32), 32, 0, stream>>>(
      XW, WP, proj_b, CH, 1, nullptr, nullptr, nullptr, x, Y, nullptr, nullptr, nullptr);
  // 8) LN2 -> f16
  ln_kernel<<<TTOT / 4, 128, 0, stream>>>(Y, g2, b2, H2, 0);
  // 9) fc1 GEMM + GELU
  gemm_wmma_kernel<<<dim3(MLPD / 48, TTOT / 32), 32, 0, stream>>>(
      H2, W1, fc1_b, CH, 2, nullptr, nullptr, nullptr, nullptr, nullptr, G, nullptr, nullptr);
  // 10) fc2 GEMM + residual -> output
  gemm_wmma_kernel<<<dim3(CH / 48, TTOT / 32), 32, 0, stream>>>(
      G, W2, fc2_b, MLPD, 3, nullptr, nullptr, nullptr, nullptr, nullptr, nullptr, Y, out);
}
